// DeformSimpleBottleneck_25357486916158
// MI455X (gfx1250) — matrix operand order repaired and checked by reading, not compiled, not run on hardware
//
#include <hip/hip_runtime.h>
#include <hip/hip_bf16.h>

typedef __bf16 bf16_t;
typedef __attribute__((ext_vector_type(16))) __bf16 v16bf;
typedef __attribute__((ext_vector_type(8)))  __bf16 v8bf;
typedef __attribute__((ext_vector_type(4)))  __bf16 v4bf;
typedef __attribute__((ext_vector_type(8)))  float  v8f;
typedef __attribute__((ext_vector_type(4)))  int    v4i;
typedef __attribute__((address_space(1))) v4i* gv4i_ptr;   // global v4i*
typedef __attribute__((address_space(3))) v4i* lv4i_ptr;   // LDS v4i*

#define EPS 1e-5f
#define Bsz 4
#define Hd  64
#define Wdm 64
#define C0  256          // planes
#define C1  512          // width
#define Gd  2
#define Kt  9
#define DIL 2
#define Hp  60
#define Wp  60
#define NROW (Bsz*Hp*Wp)         // 14400
#define CG   (C1/Gd)             // 256
#define KDEF (Gd*Kt*CG)          // 4608
#define KOFF (Kt*C1)             // 4608
#define OFFCH 54
#define OFFPAD 64
#define KC   128                 // K-chunk staged in LDS for the big GEMM
#define NSTG (KDEF / KC)         // 36

// ---------- async-LDS path availability ----------
#if __has_builtin(__builtin_amdgcn_global_load_async_to_lds_b128)
#define USE_ASYNC_LDS 1
#else
#define USE_ASYNC_LDS 0
#endif

#if __has_builtin(__builtin_amdgcn_s_wait_asynccnt)
#define WAIT_ASYNC(n) __builtin_amdgcn_s_wait_asynccnt(n)
#else
#define WAIT_ASYNC(n) asm volatile("s_wait_asynccnt %0" ::"i"(n) : "memory")
#endif

// ---------- workspace layout (bytes) ----------
constexpr size_t al256(size_t x) { return (x + 255) & ~(size_t)255; }
constexpr size_t WS_Y     = 0;                                                   // f32 [B*H*W, C1]
constexpr size_t WS_OFF   = al256(WS_Y     + (size_t)Bsz*Hd*Wdm*C1*4);           // f32 [NROW, 54]
constexpr size_t WS_A     = al256(WS_OFF   + (size_t)NROW*OFFCH*4);              // bf16 [NROW, KDEF]
constexpr size_t WS_Z     = al256(WS_A     + (size_t)NROW*KDEF*2);               // f32 [NROW, C1]
constexpr size_t WS_WT1   = al256(WS_Z     + (size_t)NROW*C1*4);                 // bf16 [C1, C0]
constexpr size_t WS_B1    = al256(WS_WT1   + (size_t)C1*C0*2);                   // f32 [C1]
constexpr size_t WS_WTOFF = al256(WS_B1    + (size_t)C1*4);                      // bf16 [OFFPAD, KOFF]
constexpr size_t WS_WD    = al256(WS_WTOFF + (size_t)OFFPAD*KOFF*2);             // bf16 [C1, KDEF]
constexpr size_t WS_B2    = al256(WS_WD    + (size_t)C1*KDEF*2);                 // f32 [C1]
constexpr size_t WS_W3T   = al256(WS_B2    + (size_t)C1*4);                      // bf16 [C0, C1]
constexpr size_t WS_B3    = al256(WS_W3T   + (size_t)C0*C1*2);                   // f32 [C0]

// ---------- fragment helpers (CDNA5 WMMA 16x16x32 bf16 layouts) ----------
__device__ __forceinline__ v8f wmma_bf16(v16bf a, v16bf b, v8f c) {
  return __builtin_amdgcn_wmma_f32_16x16x32_bf16(false, a, false, b, (short)0, c, false, false);
}

__device__ __forceinline__ v16bf mk_a_f32(const float* p0, const float* p1) {
  float4 a0 = *(const float4*)(p0);
  float4 a1 = *(const float4*)(p0 + 4);
  float4 b0 = *(const float4*)(p1);
  float4 b1 = *(const float4*)(p1 + 4);
  v16bf r;
  r[0]=(__bf16)a0.x; r[1]=(__bf16)a0.y; r[2]=(__bf16)a0.z; r[3]=(__bf16)a0.w;
  r[4]=(__bf16)a1.x; r[5]=(__bf16)a1.y; r[6]=(__bf16)a1.z; r[7]=(__bf16)a1.w;
  r[8]=(__bf16)b0.x; r[9]=(__bf16)b0.y; r[10]=(__bf16)b0.z; r[11]=(__bf16)b0.w;
  r[12]=(__bf16)b1.x; r[13]=(__bf16)b1.y; r[14]=(__bf16)b1.z; r[15]=(__bf16)b1.w;
  return r;
}

__device__ __forceinline__ v16bf mk_a_bf16(const bf16_t* p0, const bf16_t* p1) {
  v8bf lo = *(const v8bf*)p0;
  v8bf hi = *(const v8bf*)p1;
  v16bf r;
#pragma unroll
  for (int i = 0; i < 8; ++i) { r[i] = lo[i]; r[i+8] = hi[i]; }
  return r;
}

// ---------- weight prep (BN folded, transposed [N][K] bf16) ----------
__global__ __launch_bounds__(256) void k_prep_w1(const float* __restrict__ w1,
    const float* __restrict__ s, const float* __restrict__ b,
    const float* __restrict__ m, const float* __restrict__ v,
    bf16_t* __restrict__ Wt, float* __restrict__ beta) {
  int t = blockIdx.x * blockDim.x + threadIdx.x;
  if (t >= C1 * C0) return;
  int n = t / C0, c = t % C0;
  float alpha = s[n] * rsqrtf(v[n] + EPS);
  Wt[(size_t)n * C0 + c] = (__bf16)(w1[(size_t)c * C1 + n] * alpha);
  if (c == 0) beta[n] = b[n] - m[n] * alpha;
}

__global__ __launch_bounds__(256) void k_prep_woff(const float* __restrict__ w_off,
    bf16_t* __restrict__ Wt) {
  int t = blockIdx.x * blockDim.x + threadIdx.x;
  if (t >= OFFPAD * KOFF) return;
  int n = t / KOFF, r = t % KOFF;
  int tap = r / C1, c = r % C1;
  float val = (n < OFFCH) ? w_off[((size_t)tap * C1 + c) * OFFCH + n] : 0.f;
  Wt[(size_t)n * KOFF + r] = (__bf16)val;
}

__global__ __launch_bounds__(256) void k_prep_wd(const float* __restrict__ w_d,
    const float* __restrict__ s, const float* __restrict__ b,
    const float* __restrict__ m, const float* __restrict__ v,
    bf16_t* __restrict__ Wt, float* __restrict__ beta) {
  int t = blockIdx.x * blockDim.x + threadIdx.x;
  if (t >= C1 * KDEF) return;
  int f = t / KDEF, r = t % KDEF;
  int g = r / (Kt * CG), rem = r % (Kt * CG);
  int k = rem / CG, c = rem % CG;
  float alpha = s[f] * rsqrtf(v[f] + EPS);
  Wt[(size_t)f * KDEF + r] = (__bf16)(w_d[((size_t)k * C1 + (g * CG + c)) * C1 + f] * alpha);
  if (r == 0) beta[f] = b[f] - m[f] * alpha;
}

__global__ __launch_bounds__(256) void k_prep_w3(const float* __restrict__ w3,
    const float* __restrict__ s, const float* __restrict__ b,
    const float* __restrict__ m, const float* __restrict__ v,
    bf16_t* __restrict__ Wt, float* __restrict__ beta) {
  int t = blockIdx.x * blockDim.x + threadIdx.x;
  if (t >= C0 * C1) return;
  int f = t / C1, c = t % C1;
  float alpha = s[f] * rsqrtf(v[f] + EPS);
  Wt[(size_t)f * C1 + c] = (__bf16)(w3[(size_t)c * C0 + f] * alpha);
  if (c == 0) beta[f] = b[f] - m[f] * alpha;
}

// ---------- GEMM 1: y = relu(x @ W1' + beta1), M=16384 K=256 N=512 ----------
__global__ __launch_bounds__(128) void k_gemm1(const float* __restrict__ x,
    const bf16_t* __restrict__ Wt, const float* __restrict__ beta,
    float* __restrict__ y) {
  const int lane = threadIdx.x & 31;
  const int wave = (blockIdx.x * blockDim.x + threadIdx.x) >> 5;
  const int NW = C1 / 64;
  const int mt = wave / NW, nw = wave % NW;
  if (mt >= (Bsz * Hd * Wdm) / 16) return;
  const int m0 = mt * 16, n0 = nw * 64;
  const int lrow = lane & 15, half = lane >> 4;
  v8f acc[4] = {};
  const float* arow = x + (size_t)(m0 + lrow) * C0;
  for (int kk = 0; kk < C0; kk += 32) {
    v16bf a = mk_a_f32(arow + kk + half * 16, arow + kk + half * 16 + 8);
#pragma unroll
    for (int ns = 0; ns < 4; ++ns) {
      const bf16_t* bp = Wt + (size_t)(n0 + ns * 16 + lrow) * C0 + kk + half * 16;
      acc[ns] = wmma_bf16(a, *(const v16bf*)bp, acc[ns]);
    }
  }
#pragma unroll
  for (int ns = 0; ns < 4; ++ns) {
    int col = n0 + ns * 16 + lrow;
    float bb = beta[col];
#pragma unroll
    for (int vv = 0; vv < 8; ++vv) {
      int mrow = m0 + half * 8 + vv;
      y[(size_t)mrow * C1 + col] = fmaxf(acc[ns][vv] + bb, 0.f);
    }
  }
}

// ---------- GEMM 2: offset conv (3x3 dil=2 VALID), M=14400 K=4608 N=54(pad64) ----------
__global__ __launch_bounds__(128) void k_gemm_off(const float* __restrict__ y,
    const bf16_t* __restrict__ Wt, const float* __restrict__ b_off,
    float* __restrict__ off) {
  const int lane = threadIdx.x & 31;
  const int wave = (blockIdx.x * blockDim.x + threadIdx.x) >> 5;
  if (wave >= NROW / 16) return;
  const int m0 = wave * 16;
  const int lrow = lane & 15, half = lane >> 4;
  const int row = m0 + lrow;
  const int b = row / (Hp * Wp), rem = row % (Hp * Wp);
  const int i = rem / Wp, j = rem % Wp;
  v8f acc[4] = {};
  for (int kk = 0; kk < KOFF; kk += 32) {
    const int tap = kk >> 9, kloc = kk & 511;
    const int ky = tap / 3, kx = tap % 3;
    const float* arow = y + (((size_t)(b * Hd + i + ky * DIL)) * Wdm + (j + kx * DIL)) * C1 + kloc;
    v16bf a = mk_a_f32(arow + half * 16, arow + half * 16 + 8);
#pragma unroll
    for (int ns = 0; ns < 4; ++ns) {
      const bf16_t* bp = Wt + (size_t)(ns * 16 + lrow) * KOFF + kk + half * 16;
      acc[ns] = wmma_bf16(a, *(const v16bf*)bp, acc[ns]);
    }
  }
#pragma unroll
  for (int ns = 0; ns < 4; ++ns) {
    int col = ns * 16 + lrow;
    if (col < OFFCH) {
      float bb = b_off[col];
#pragma unroll
      for (int vv = 0; vv < 8; ++vv) {
        int mrow = m0 + half * 8 + vv;
        off[(size_t)mrow * OFFCH + col] = acc[ns][vv] + bb;
      }
    }
  }
}

// ---------- gather: modulated bilinear sample -> bf16 A-matrix [NROW, KDEF] ----------
__global__ __launch_bounds__(256) void k_gather(const float* __restrict__ y,
    const float* __restrict__ off, bf16_t* __restrict__ A) {
  int t = blockIdx.x * blockDim.x + threadIdx.x;
  if (t >= NROW * Gd * Kt) return;
  const int row = t / (Gd * Kt), gk = t % (Gd * Kt);
  const int g = gk / Kt, k = gk % Kt;
  const int b = row / (Hp * Wp), rem = row % (Hp * Wp);
  const int i = rem / Wp, j = rem % Wp;
  const float* orow = off + (size_t)row * OFFCH;
  const float dy = orow[(g * Kt + k) * 2 + 0];
  const float dx = orow[(g * Kt + k) * 2 + 1];
  const float msk = 1.f / (1.f + __expf(-orow[2 * Gd * Kt + g * Kt + k]));
  const float py = (float)i + (float)((k / 3) * DIL) + dy;
  const float px = (float)j + (float)((k % 3) * DIL) + dx;
  const float fy0 = floorf(py), fx0 = floorf(px);
  const int y0 = (int)fy0, x0 = (int)fx0;
  const float wy1 = py - fy0, wx1 = px - fx0;
  const float w00 = (1.f - wy1) * (1.f - wx1) * msk;
  const float w01 = (1.f - wy1) * wx1 * msk;
  const float w10 = wy1 * (1.f - wx1) * msk;
  const float w11 = wy1 * wx1 * msk;
  const bool okY0 = (y0 >= 0) & (y0 < Hd), okY1 = (y0 + 1 >= 0) & (y0 + 1 < Hd);
  const bool okX0 = (x0 >= 0) & (x0 < Wdm), okX1 = (x0 + 1 >= 0) & (x0 + 1 < Wdm);
  const bool v00 = okY0 & okX0, v01 = okY0 & okX1, v10 = okY1 & okX0, v11 = okY1 & okX1;
  const int yc0 = min(max(y0, 0), Hd - 1), yc1 = min(max(y0 + 1, 0), Hd - 1);
  const int xc0 = min(max(x0, 0), Wdm - 1), xc1 = min(max(x0 + 1, 0), Wdm - 1);
  const float* p00 = y + (((size_t)(b * Hd + yc0)) * Wdm + xc0) * C1 + g * CG;
  const float* p01 = y + (((size_t)(b * Hd + yc0)) * Wdm + xc1) * C1 + g * CG;
  const float* p10 = y + (((size_t)(b * Hd + yc1)) * Wdm + xc0) * C1 + g * CG;
  const float* p11 = y + (((size_t)(b * Hd + yc1)) * Wdm + xc1) * C1 + g * CG;
  bf16_t* dst = A + (size_t)row * KDEF + (size_t)(g * Kt + k) * CG;
  for (int c = 0; c < CG; c += 4) {
    float sx = 0.f, sy = 0.f, sz = 0.f, sw = 0.f;
    if (v00) { float4 q = *(const float4*)(p00 + c); sx += w00*q.x; sy += w00*q.y; sz += w00*q.z; sw += w00*q.w; }
    if (v01) { float4 q = *(const float4*)(p01 + c); sx += w01*q.x; sy += w01*q.y; sz += w01*q.z; sw += w01*q.w; }
    if (v10) { float4 q = *(const float4*)(p10 + c); sx += w10*q.x; sy += w10*q.y; sz += w10*q.z; sw += w10*q.w; }
    if (v11) { float4 q = *(const float4*)(p11 + c); sx += w11*q.x; sy += w11*q.y; sz += w11*q.z; sw += w11*q.w; }
    v4bf o; o[0] = (__bf16)sx; o[1] = (__bf16)sy; o[2] = (__bf16)sz; o[3] = (__bf16)sw;
    *(v4bf*)(dst + c) = o;
  }
}

// ---------- GEMM 4 (dominant): z = relu(A @ Wd' + beta2), M=14400 K=4608 N=512 ----
// Block = 4 waves sharing a 64-wide N panel; B staged in LDS (double buffered),
// async global->LDS when available, ds_load_b128 fragment reads.
__global__ __launch_bounds__(128) void k_gemm_def(const bf16_t* __restrict__ A,
    const bf16_t* __restrict__ Wt, const float* __restrict__ beta,
    float* __restrict__ z) {
  __shared__ __align__(32) bf16_t sB[2][64 * KC];   // 2 x 16 KB
  const int tid  = threadIdx.x;
  const int lane = tid & 31;
  const int wave = tid >> 5;
  const int m0 = blockIdx.x * 64 + wave * 16;       // 225 M-blocks of 64 rows
  const int n0 = blockIdx.y * 64;                   // 8 N-blocks of 64 cols
  const int lrow = lane & 15, half = lane >> 4;

  // each thread stages 8 x 16B chunks of the 64x128 bf16 B panel per stage
  auto issue = [&](int buf, int kk) {
#pragma unroll
    for (int q = 0; q < 8; ++q) {
      const int ch = tid + q * 128;                 // 0..1023
      const int n  = ch >> 4;                       // panel row
      const int c8 = (ch & 15) << 3;                // col start (8 bf16 = 16B)
      const bf16_t* g = Wt + (size_t)(n0 + n) * KDEF + kk + c8;
      bf16_t* l = &sB[buf][ch * 8];
#if USE_ASYNC_LDS
      __builtin_amdgcn_global_load_async_to_lds_b128(
          (gv4i_ptr)(void*)g, (lv4i_ptr)(void*)l, 0, 0);
#else
      *(uint4*)l = *(const uint4*)g;
#endif
    }
  };

  v8f acc[4] = {};
  const bf16_t* arow = A + (size_t)(m0 + lrow) * KDEF;

  issue(0, 0);
  for (int s = 0; s < NSTG; ++s) {
    if (s + 1 < NSTG) {
      issue((s + 1) & 1, (s + 1) * KC);
#if USE_ASYNC_LDS
      WAIT_ASYNC(8);          // current stage's 8 chunks landed (in-order)
#endif
    } else {
#if USE_ASYNC_LDS
      WAIT_ASYNC(0);
#endif
    }
    __syncthreads();
    const bf16_t* bufp = sB[s & 1];
    const int kk = s * KC;
#pragma unroll
    for (int ks = 0; ks < KC / 32; ++ks) {
      const bf16_t* ap = arow + kk + ks * 32;
      if (ks == 0) __builtin_prefetch(ap + 2 * KC, 0, 0);
      v16bf a = mk_a_bf16(ap + half * 16, ap + half * 16 + 8);
#pragma unroll
      for (int ns = 0; ns < 4; ++ns) {
        const bf16_t* bp = bufp + (ns * 16 + lrow) * KC + ks * 32 + half * 16;
        acc[ns] = wmma_bf16(a, *(const v16bf*)bp, acc[ns]);
      }
    }
    __syncthreads();
  }

#pragma unroll
  for (int ns = 0; ns < 4; ++ns) {
    int col = n0 + ns * 16 + lrow;
    float bb = beta[col];
#pragma unroll
    for (int vv = 0; vv < 8; ++vv) {
      int mrow = m0 + half * 8 + vv;
      z[(size_t)mrow * C1 + col] = fmaxf(acc[ns][vv] + bb, 0.f);
    }
  }
}

// ---------- GEMM 5: out = relu(z @ W3' + beta3 + identity), padded store ----------
__global__ __launch_bounds__(128) void k_gemm_out(const float* __restrict__ z,
    const bf16_t* __restrict__ Wt, const float* __restrict__ beta,
    const float* __restrict__ x, float* __restrict__ out) {
  const int lane = threadIdx.x & 31;
  const int wave = (blockIdx.x * blockDim.x + threadIdx.x) >> 5;
  const int NW = C0 / 64;
  const int mt = wave / NW, nw = wave % NW;
  if (mt >= NROW / 16) return;
  const int m0 = mt * 16, n0 = nw * 64;
  const int lrow = lane & 15, half = lane >> 4;
  v8f acc[4] = {};
  const float* arow = z + (size_t)(m0 + lrow) * C1;
  for (int kk = 0; kk < C1; kk += 32) {
    v16bf a = mk_a_f32(arow + kk + half * 16, arow + kk + half * 16 + 8);
#pragma unroll
    for (int ns = 0; ns < 4; ++ns) {
      const bf16_t* bp = Wt + (size_t)(n0 + ns * 16 + lrow) * C1 + kk + half * 16;
      acc[ns] = wmma_bf16(a, *(const v16bf*)bp, acc[ns]);
    }
  }
#pragma unroll
  for (int ns = 0; ns < 4; ++ns) {
    int col = n0 + ns * 16 + lrow;
    float bb = beta[col];
#pragma unroll
    for (int vv = 0; vv < 8; ++vv) {
      int mrow = m0 + half * 8 + vv;
      int b = mrow / (Hp * Wp), rem = mrow % (Hp * Wp);
      int i = rem / Wp, j = rem % Wp;
      size_t oi = (((size_t)(b * Hd + i + DIL)) * Wdm + (j + DIL)) * C0 + col;
      out[oi] = fmaxf(acc[ns][vv] + bb + x[oi], 0.f);
    }
  }
}

// ---------- border ring: out = relu(identity) ----------
__global__ __launch_bounds__(256) void k_border(const float* __restrict__ x,
    float* __restrict__ out) {
  int t = blockIdx.x * blockDim.x + threadIdx.x;
  if (t >= Bsz * Hd * Wdm * C0) return;
  int rest = t / C0;
  int j = rest % Wdm, i = (rest / Wdm) % Hd;
  if (i < DIL || i >= Hd - DIL || j < DIL || j >= Wdm - DIL)
    out[t] = fmaxf(x[t], 0.f);
}

extern "C" void kernel_launch(void* const* d_in, const int* in_sizes, int n_in,
                              void* d_out, int out_size, void* d_ws, size_t ws_size,
                              hipStream_t stream) {
  (void)in_sizes; (void)n_in; (void)out_size; (void)ws_size;
  const float* x     = (const float*)d_in[0];
  const float* w1    = (const float*)d_in[1];
  const float* s1    = (const float*)d_in[2];
  const float* b1    = (const float*)d_in[3];
  const float* m1    = (const float*)d_in[4];
  const float* v1    = (const float*)d_in[5];
  const float* w_off = (const float*)d_in[6];
  const float* b_off = (const float*)d_in[7];
  const float* w_d   = (const float*)d_in[8];
  const float* s2    = (const float*)d_in[9];
  const float* b2    = (const float*)d_in[10];
  const float* m2    = (const float*)d_in[11];
  const float* v2    = (const float*)d_in[12];
  const float* w3    = (const float*)d_in[13];
  const float* s3    = (const float*)d_in[14];
  const float* b3    = (const float*)d_in[15];
  const float* m3    = (const float*)d_in[16];
  const float* v3    = (const float*)d_in[17];
  float* out = (float*)d_out;

  char* ws = (char*)d_ws;
  float*  y     = (float*) (ws + WS_Y);
  float*  offb  = (float*) (ws + WS_OFF);
  bf16_t* Amat  = (bf16_t*)(ws + WS_A);
  float*  z     = (float*) (ws + WS_Z);
  bf16_t* Wt1   = (bf16_t*)(ws + WS_WT1);
  float*  beta1 = (float*) (ws + WS_B1);
  bf16_t* Wtoff = (bf16_t*)(ws + WS_WTOFF);
  bf16_t* Wdt   = (bf16_t*)(ws + WS_WD);
  float*  beta2 = (float*) (ws + WS_B2);
  bf16_t* W3t   = (bf16_t*)(ws + WS_W3T);
  float*  beta3 = (float*) (ws + WS_B3);

  // weight prep (BN folded)
  k_prep_w1 <<<(C1*C0 + 255)/256, 256, 0, stream>>>(w1, s1, b1, m1, v1, Wt1, beta1);
  k_prep_woff<<<(OFFPAD*KOFF + 255)/256, 256, 0, stream>>>(w_off, Wtoff);
  k_prep_wd <<<(C1*KDEF + 255)/256, 256, 0, stream>>>(w_d, s2, b2, m2, v2, Wdt, beta2);
  k_prep_w3 <<<(C0*C1 + 255)/256, 256, 0, stream>>>(w3, s3, b3, m3, v3, W3t, beta3);

  // 1) y = relu(BN1(x @ w1))          16384x512, K=256
  {
    int waves = ((Bsz*Hd*Wdm)/16) * (C1/64);      // 8192
    k_gemm1<<<waves/4, 128, 0, stream>>>(x, Wt1, beta1, y);
  }
  // 2) off = offset conv              14400x54, K=4608
  {
    int waves = NROW/16;                          // 900
    k_gemm_off<<<(waves + 3)/4, 128, 0, stream>>>(y, Wtoff, b_off, offb);
  }
  // 3) bilinear modulated gather -> bf16 A-matrix
  k_gather<<<(NROW*Gd*Kt + 255)/256, 256, 0, stream>>>(y, offb, Amat);
  // 4) z = relu(BN2(A @ wk))          14400x512, K=4608 (dominant, LDS-staged)
  {
    dim3 grid(NROW/64, C1/64);                    // 225 x 8
    k_gemm_def<<<grid, 128, 0, stream>>>(Amat, Wdt, beta2, z);
  }
  // 5) out interior = relu(BN3(z @ w3) + identity), with pad offset
  {
    int waves = (NROW/16) * (C0/64);              // 3600
    k_gemm_out<<<waves/4, 128, 0, stream>>>(z, W3t, beta3, x, out);
  }
  // 6) border ring = relu(identity)
  k_border<<<(Bsz*Hd*Wdm*C0 + 255)/256, 256, 0, stream>>>(x, out);
}
